// Final_ResGRU_26929444946709
// MI455X (gfx1250) — matrix operand order repaired and checked by reading, not compiled, not run on hardware
//
#include <hip/hip_runtime.h>
#include <math.h>
#include <stdint.h>

typedef __bf16 bf16_t;
typedef __attribute__((ext_vector_type(16))) __bf16 v16bf;
typedef __attribute__((ext_vector_type(8)))  float  v8f;

#define B_SZ 1024
#define T_SZ 100
#define BT   (B_SZ * T_SZ)

// --- CDNA5 async LDS load path (guarded; falls back to sync staging) -------
#if __has_builtin(__builtin_amdgcn_global_load_async_to_lds_b128)
#define HAVE_ASYNC 1
#else
#define HAVE_ASYNC 0
#endif

typedef __attribute__((ext_vector_type(4))) int i4v;
typedef __attribute__((address_space(1))) i4v g_i4;   // global addrspace int4
typedef __attribute__((address_space(3))) i4v l_i4;   // LDS addrspace int4
// global flat address == as(1) address numerically; LDS generic pointer's
// low 32 bits == LDS byte offset (flat aperture rule)
#define TO_GLB(p) ((g_i4*)(uintptr_t)(p))
#define TO_LDS(p) ((l_i4*)(unsigned)(uintptr_t)(p))

#if HAVE_ASYNC
#if __has_builtin(__builtin_amdgcn_s_wait_asynccnt)
#define WAIT_ASYNC(n) __builtin_amdgcn_s_wait_asynccnt(n)
#else
#define WAIT_ASYNC(n) asm volatile("s_wait_asynccnt %0" ::"i"(n) : "memory")
#endif
#else
#define WAIT_ASYNC(n)
#endif

// ---------------------------------------------------------------------------
// f32 -> bf16 elementwise (weight conversion)
// ---------------------------------------------------------------------------
__global__ void k_f32_to_bf16(const float* __restrict__ x, bf16_t* __restrict__ y, int n) {
    int i = blockIdx.x * blockDim.x + threadIdx.x;
    if (i < n) y[i] = (bf16_t)x[i];
}

// ---------------------------------------------------------------------------
// LayerNorm over last dim D (multiple of 32), one wave per row, bf16 out
// ---------------------------------------------------------------------------
__global__ __launch_bounds__(256) void k_layernorm_bf16(
    const float* __restrict__ x, const float* __restrict__ g, const float* __restrict__ b,
    bf16_t* __restrict__ y, int rows, int D) {
    int row  = blockIdx.x * (blockDim.x >> 5) + (threadIdx.x >> 5);
    int lane = threadIdx.x & 31;
    if (row >= rows) return;
    const float* xr = x + (size_t)row * D;
    int per = D >> 5;
    float vals[8];
    float s = 0.f;
    for (int i = 0; i < per; ++i) { vals[i] = xr[lane + 32 * i]; s += vals[i]; }
    for (int o = 16; o > 0; o >>= 1) s += __shfl_xor(s, o, 32);
    float mu = s / (float)D;
    float v = 0.f;
    for (int i = 0; i < per; ++i) { float d = vals[i] - mu; v += d * d; }
    for (int o = 16; o > 0; o >>= 1) v += __shfl_xor(v, o, 32);
    float rstd = rsqrtf(v / (float)D + 1e-5f);
    bf16_t* yr = y + (size_t)row * D;
    for (int i = 0; i < per; ++i) {
        int c = lane + 32 * i;
        yr[c] = (bf16_t)((vals[i] - mu) * rstd * g[c] + b[c]);
    }
}

// ---------------------------------------------------------------------------
// Tiled bf16 WMMA GEMM: C[M,N] = A[M,K] * W[N,K]^T + bias[N]
// TBMxTBN block tile, BK=32, 8 waves; per-wave MTxNT 16x16 WMMA tiles.
// Double-buffered LDS fed by async global->LDS loads (ASYNCcnt) when available.
// ---------------------------------------------------------------------------
#define GBK 32
#define LST 40   // LDS row stride in bf16 elems (80B: 16B aligned + conflict pad)

template <int TBM, int TBN, int MT, int NT, bool OUTF32>
__global__ __launch_bounds__(256) void k_gemm(
    const bf16_t* __restrict__ A, const bf16_t* __restrict__ W,
    const float* __restrict__ bias, void* __restrict__ Cv,
    int M, int N, int K) {
    constexpr int MW  = TBM / (16 * MT);   // waves along M
    constexpr int NW  = TBN / (16 * NT);   // waves along N
    static_assert(MW * NW == 8, "8 waves per block");
    constexpr int API = (TBM * 4) / 256;   // A 16B-chunks per thread / async instrs per wave
    constexpr int WPI = (TBN * 4) / 256;
    constexpr int NPW = API + WPI;         // async instrs per wave per tile

    __shared__ bf16_t As[2][TBM * LST];
    __shared__ bf16_t Ws[2][TBN * LST];

    int tid  = threadIdx.x;
    int bm   = blockIdx.y * TBM;
    int bn   = blockIdx.x * TBN;
    int wave = tid >> 5;
    int lane = tid & 31;
    int half = lane >> 4;
    int l16  = lane & 15;
    int wm   = (wave % MW) * (MT * 16);
    int wn   = (wave / MW) * (NT * 16);

    v8f acc[MT][NT] = {};

    auto stage = [&](int buf, int k0) {
#if HAVE_ASYNC
#pragma unroll
        for (int i = 0; i < API; ++i) {
            int c = (wave * API + i) * 32 + lane;
            int r = c >> 2, cc = c & 3;
            __builtin_amdgcn_global_load_async_to_lds_b128(
                TO_GLB(A + (size_t)(bm + r) * K + k0 + cc * 8),
                TO_LDS(&As[buf][r * LST + cc * 8]), 0, 0);
        }
#pragma unroll
        for (int i = 0; i < WPI; ++i) {
            int c = (wave * WPI + i) * 32 + lane;
            int r = c >> 2, cc = c & 3;
            __builtin_amdgcn_global_load_async_to_lds_b128(
                TO_GLB(W + (size_t)(bn + r) * K + k0 + cc * 8),
                TO_LDS(&Ws[buf][r * LST + cc * 8]), 0, 0);
        }
#else
#pragma unroll
        for (int i = 0; i < API; ++i) {
            int c = (wave * API + i) * 32 + lane;
            int r = c >> 2, cc = c & 3;
            *(uint4*)(&As[buf][r * LST + cc * 8]) =
                *(const uint4*)(A + (size_t)(bm + r) * K + k0 + cc * 8);
        }
#pragma unroll
        for (int i = 0; i < WPI; ++i) {
            int c = (wave * WPI + i) * 32 + lane;
            int r = c >> 2, cc = c & 3;
            *(uint4*)(&Ws[buf][r * LST + cc * 8]) =
                *(const uint4*)(W + (size_t)(bn + r) * K + k0 + cc * 8);
        }
#endif
    };

    int ntiles = K / GBK;
    stage(0, 0);
    for (int it = 0; it < ntiles; ++it) {
        int cur = it & 1;
        if (it + 1 < ntiles) {
            stage(1 - cur, (it + 1) * GBK);   // prefetch next tile into other buffer
            WAIT_ASYNC(NPW);                  // retire current tile's loads (in-order)
        } else {
            WAIT_ASYNC(0);
        }
        __syncthreads();                      // all waves' slices visible

        v16bf afrag[MT], bfrag[NT];
#pragma unroll
        for (int mt = 0; mt < MT; ++mt) {
            int r = wm + mt * 16 + l16;
            union { v16bf v; uint4 q[2]; } u;
            // CDNA5 16-bit A layout: e0..7 -> k=8*half+e ; e8..15 -> k=16+8*half+(e-8)
            u.q[0] = *(const uint4*)(&As[cur][r * LST + 8 * half]);
            u.q[1] = *(const uint4*)(&As[cur][r * LST + 16 + 8 * half]);
            afrag[mt] = u.v;
        }
#pragma unroll
        for (int nt = 0; nt < NT; ++nt) {
            int r = wn + nt * 16 + l16;
            union { v16bf v; uint4 q[2]; } u;
            // CDNA5 16-bit B layout: e0..7 -> k=8*half+e ; e8..15 -> k=16+8*half+(e-8)
            u.q[0] = *(const uint4*)(&Ws[cur][r * LST + 8 * half]);
            u.q[1] = *(const uint4*)(&Ws[cur][r * LST + 16 + 8 * half]);
            bfrag[nt] = u.v;
        }
#pragma unroll
        for (int mt = 0; mt < MT; ++mt)
#pragma unroll
            for (int nt = 0; nt < NT; ++nt)
                acc[mt][nt] = __builtin_amdgcn_wmma_f32_16x16x32_bf16(
                    false, afrag[mt], false, bfrag[nt], (short)0, acc[mt][nt],
                    false, false);
        __syncthreads();                      // done reading cur before it is re-staged
    }

#pragma unroll
    for (int mt = 0; mt < MT; ++mt)
#pragma unroll
        for (int nt = 0; nt < NT; ++nt) {
            int col  = bn + wn + nt * 16 + l16;
            float bv = bias[col];
#pragma unroll
            for (int r = 0; r < 8; ++r) {
                int row = bm + wm + mt * 16 + r + 8 * half;
                float v = acc[mt][nt][r] + bv;
                if (OUTF32) ((float*)Cv)[(size_t)row * N + col] = v;
                else        ((bf16_t*)Cv)[(size_t)row * N + col] = (bf16_t)v;
            }
        }
}

// ---------------------------------------------------------------------------
// GRU gate math for one timestep (f32 hidden master + bf16 mirror)
// ---------------------------------------------------------------------------
__global__ __launch_bounds__(256) void k_gru_gates(
    const bf16_t* __restrict__ xg, const float* __restrict__ gh,
    float* __restrict__ hf, bf16_t* __restrict__ hb,
    float* __restrict__ out, const int* __restrict__ lens,
    int t, int T, int H, int outStride, int outColOff, int residual) {
    int idx = blockIdx.x * blockDim.x + threadIdx.x;  // exactly B*H threads
    int b = idx / H, j = idx - b * H;
    bool m = t < lens[b];
    size_t xb = ((size_t)b * T + t) * (size_t)(3 * H);
    float ir  = (float)xg[xb + j];
    float iz  = (float)xg[xb + H + j];
    float inn = (float)xg[xb + 2 * H + j];
    size_t gb = (size_t)b * (size_t)(3 * H);
    float hr = gh[gb + j], hz = gh[gb + H + j], hn = gh[gb + 2 * H + j];
    float r = 1.f / (1.f + __expf(-(ir + hr)));
    float z = 1.f / (1.f + __expf(-(iz + hz)));
    float n = tanhf(inn + r * hn);
    float hp   = hf[(size_t)b * H + j];
    float hnew = m ? ((1.f - z) * n + z * hp) : hp;
    hf[(size_t)b * H + j] = hnew;
    hb[(size_t)b * H + j] = (bf16_t)hnew;
    float ov = m ? hnew : 0.f;
    size_t oi = ((size_t)b * T + t) * (size_t)outStride + outColOff + j;
    out[oi] = residual ? (out[oi] + ov) : ov;
}

// ---------------------------------------------------------------------------
// Reverse-gather of bf16 rows: y[b,t,:] = x[b, clip(len-1-t,0), :]
// ---------------------------------------------------------------------------
__global__ void k_gather_rev_bf16(const bf16_t* __restrict__ x, bf16_t* __restrict__ y,
                                  const int* __restrict__ lens, int T, int D) {
    int b = blockIdx.x, t = blockIdx.y;
    int r = lens[b] - 1 - t; if (r < 0) r = 0;
    const bf16_t* src = x + ((size_t)b * T + r) * D;
    bf16_t*       dst = y + ((size_t)b * T + t) * D;
    for (int j = threadIdx.x; j < D; j += blockDim.x) dst[j] = src[j];
}

// ---------------------------------------------------------------------------
// Un-reverse + mask into H1 columns
// ---------------------------------------------------------------------------
__global__ void k_unrev(const float* __restrict__ obrev, float* __restrict__ h1,
                        const int* __restrict__ lens, int T, int H, int stride, int colOff) {
    int b = blockIdx.x, t = blockIdx.y;
    int len = lens[b];
    bool m = t < len;
    int r = len - 1 - t; if (r < 0) r = 0;
    const float* src = obrev + ((size_t)b * T + r) * H;
    float*       dst = h1 + ((size_t)b * T + t) * stride + colOff;
    for (int j = threadIdx.x; j < H; j += blockDim.x) dst[j] = m ? src[j] : 0.f;
}

// ---------------------------------------------------------------------------
// Pooling: last valid row + masked max over time
// ---------------------------------------------------------------------------
__global__ __launch_bounds__(256) void k_pool(
    const float* __restrict__ h, const int* __restrict__ lens,
    float* __restrict__ mlpin, int T, int H, int colOff, int mlpStride) {
    int idx = blockIdx.x * blockDim.x + threadIdx.x;  // B*H threads
    int b = idx / H, j = idx - b * H;
    int len = lens[b];
    const float* hb = h + (size_t)b * T * H + j;
    float last = hb[(size_t)(len - 1) * H];
    float mx = -3.0e38f;
    for (int t = 0; t < len; ++t) mx = fmaxf(mx, hb[(size_t)t * H]);
    mlpin[(size_t)b * mlpStride + colOff + j]     = last;
    mlpin[(size_t)b * mlpStride + colOff + H + j] = mx;
}

// ---------------------------------------------------------------------------
// BatchNorm(eval) + LeakyReLU(0.01), bf16 out (feeds next WMMA GEMM)
// ---------------------------------------------------------------------------
__global__ __launch_bounds__(256) void k_bn_lrelu_bf16(
    const float* __restrict__ x, const float* __restrict__ g, const float* __restrict__ b,
    const float* __restrict__ m, const float* __restrict__ v,
    bf16_t* __restrict__ y, int D) {
    int idx = blockIdx.x * blockDim.x + threadIdx.x;  // rows*D threads
    int col = idx % D;
    float val = (x[idx] - m[col]) * rsqrtf(v[col] + 1e-5f) * g[col] + b[col];
    val = val > 0.f ? val : 0.01f * val;
    y[idx] = (bf16_t)val;
}

// ---------------------------------------------------------------------------
// Final 256 -> 10 head (negligible FLOPs)
// ---------------------------------------------------------------------------
__global__ void k_final(const bf16_t* __restrict__ x, const float* __restrict__ W,
                        const float* __restrict__ bias, float* __restrict__ out) {
    int b = blockIdx.x;
    int o = threadIdx.x;
    if (o >= 10) return;
    float s = bias[o];
    const bf16_t* xr = x + (size_t)b * 256;
    const float*  wr = W + (size_t)o * 256;
    for (int k = 0; k < 256; ++k) s += (float)xr[k] * wr[k];
    out[b * 10 + o] = s;
}

// ---------------------------------------------------------------------------
// Host orchestration
// ---------------------------------------------------------------------------
extern "C" void kernel_launch(void* const* d_in, const int* in_sizes, int n_in,
                              void* d_out, int out_size, void* d_ws, size_t ws_size,
                              hipStream_t stream) {
    (void)in_sizes; (void)n_in; (void)out_size; (void)ws_size;
    const float* inp[2] = { (const float*)d_in[0], (const float*)d_in[1] };
    const int*   lens   = (const int*)d_in[2];
    const int sbase[2] = { 3, 25 };

    size_t off = 0;
    auto WS = [&](size_t bytes) -> char* {
        char* p = (char*)d_ws + off;
        off += (bytes + 255) & ~(size_t)255;
        return p;
    };
    bf16_t* wb[2][8];
    const int wsz[8] = { 384*128, 384*128, 384*128, 384*128,
                         768*256, 768*256, 768*256, 768*256 };
    for (int s = 0; s < 2; ++s)
        for (int i = 0; i < 8; ++i) wb[s][i] = (bf16_t*)WS((size_t)wsz[i] * 2);
    bf16_t* mw1 = (bf16_t*)WS((size_t)1024 * 1024 * 2);
    bf16_t* mw2 = (bf16_t*)WS((size_t)512 * 1024 * 2);
    bf16_t* mw3 = (bf16_t*)WS((size_t)256 * 512 * 2);

    bf16_t* ABF  = (bf16_t*)WS((size_t)BT * 256 * 2);
    bf16_t* XG   = (bf16_t*)WS((size_t)BT * 768 * 2);
    float*  H1   = (float*) WS((size_t)BT * 256 * 4);
    char*   DIR  =          WS((size_t)BT * 128 * 4);
    float*  GH   = (float*) WS((size_t)B_SZ * 768 * 4);
    float*  HF   = (float*) WS((size_t)B_SZ * 256 * 4);
    bf16_t* HB   = (bf16_t*)WS((size_t)B_SZ * 256 * 2);
    float*  MLPI = (float*) WS((size_t)B_SZ * 1024 * 4);
    float*  MLPF = (float*) WS((size_t)B_SZ * 1024 * 4);
    bf16_t* MLPB = (bf16_t*)WS((size_t)B_SZ * 1024 * 2);

    auto cvt = [&](int idx, bf16_t* dst, int n) {
        k_f32_to_bf16<<<(n + 255) / 256, 256, 0, stream>>>((const float*)d_in[idx], dst, n);
    };
    const int wmap[8] = { 2, 3, 6, 7, 12, 13, 18, 19 };
    for (int s = 0; s < 2; ++s)
        for (int i = 0; i < 8; ++i) cvt(sbase[s] + wmap[i], wb[s][i], wsz[i]);
    cvt(51, mw1, 1024 * 1024);
    cvt(57, mw2, 512 * 1024);
    cvt(63, mw3, 256 * 512);

    // big tiles for time-parallel GEMMs, small tiles for latency-critical M=1024 GEMMs
    auto gemm_big_bf16 = [&](const bf16_t* A, const bf16_t* Wt, const float* bias,
                             bf16_t* C, int M, int N, int K) {
        k_gemm<128, 128, 4, 2, false><<<dim3(N / 128, M / 128), 256, 0, stream>>>(
            A, Wt, bias, C, M, N, K);
    };
    auto gemm_sml_f32 = [&](const bf16_t* A, const bf16_t* Wt, const float* bias,
                            float* C, int M, int N, int K) {
        k_gemm<64, 64, 1, 2, true><<<dim3(N / 64, M / 64), 256, 0, stream>>>(
            A, Wt, bias, C, M, N, K);
    };

    for (int s = 0; s < 2; ++s) {
        int bse = sbase[s];
        const float* ln1g = (const float*)d_in[bse + 0];
        const float* ln1b = (const float*)d_in[bse + 1];
        const float* b1fi = (const float*)d_in[bse + 4];
        const float* b1fh = (const float*)d_in[bse + 5];
        const float* b1bi = (const float*)d_in[bse + 8];
        const float* b1bh = (const float*)d_in[bse + 9];
        const float* ln2g = (const float*)d_in[bse + 10];
        const float* ln2b = (const float*)d_in[bse + 11];
        const float* b2i  = (const float*)d_in[bse + 14];
        const float* b2h  = (const float*)d_in[bse + 15];
        const float* ln3g = (const float*)d_in[bse + 16];
        const float* ln3b = (const float*)d_in[bse + 17];
        const float* b3i  = (const float*)d_in[bse + 20];
        const float* b3h  = (const float*)d_in[bse + 21];

        // ---- bigru layer 1 ----
        k_layernorm_bf16<<<BT / 8, 256, 0, stream>>>(inp[s], ln1g, ln1b, ABF, BT, 128);

        // forward direction
        gemm_big_bf16(ABF, wb[s][0], b1fi, XG, BT, 384, 128);
        (void)hipMemsetAsync(HF, 0, (size_t)B_SZ * 128 * 4, stream);
        (void)hipMemsetAsync(HB, 0, (size_t)B_SZ * 128 * 2, stream);
        for (int t = 0; t < T_SZ; ++t) {
            gemm_sml_f32(HB, wb[s][1], b1fh, GH, B_SZ, 384, 128);
            k_gru_gates<<<B_SZ * 128 / 256, 256, 0, stream>>>(
                XG, GH, HF, HB, H1, lens, t, T_SZ, 128, 256, 0, 0);
        }
        // backward direction (on reversed, LN'd input)
        k_gather_rev_bf16<<<dim3(B_SZ, T_SZ), 64, 0, stream>>>(
            ABF, (bf16_t*)DIR, lens, T_SZ, 128);
        gemm_big_bf16((bf16_t*)DIR, wb[s][2], b1bi, XG, BT, 384, 128);
        (void)hipMemsetAsync(HF, 0, (size_t)B_SZ * 128 * 4, stream);
        (void)hipMemsetAsync(HB, 0, (size_t)B_SZ * 128 * 2, stream);
        for (int t = 0; t < T_SZ; ++t) {
            gemm_sml_f32(HB, wb[s][3], b1bh, GH, B_SZ, 384, 128);
            k_gru_gates<<<B_SZ * 128 / 256, 256, 0, stream>>>(
                XG, GH, HF, HB, (float*)DIR, lens, t, T_SZ, 128, 128, 0, 0);
        }
        k_unrev<<<dim3(B_SZ, T_SZ), 64, 0, stream>>>(
            (float*)DIR, H1, lens, T_SZ, 128, 256, 128);

        // ---- residual gru2 ----
        k_layernorm_bf16<<<BT / 8, 256, 0, stream>>>(H1, ln2g, ln2b, ABF, BT, 256);
        gemm_big_bf16(ABF, wb[s][4], b2i, XG, BT, 768, 256);
        (void)hipMemsetAsync(HF, 0, (size_t)B_SZ * 256 * 4, stream);
        (void)hipMemsetAsync(HB, 0, (size_t)B_SZ * 256 * 2, stream);
        for (int t = 0; t < T_SZ; ++t) {
            gemm_sml_f32(HB, wb[s][5], b2h, GH, B_SZ, 768, 256);
            k_gru_gates<<<B_SZ * 256 / 256, 256, 0, stream>>>(
                XG, GH, HF, HB, H1, lens, t, T_SZ, 256, 256, 0, 1);
        }

        // ---- residual gru3 ----
        k_layernorm_bf16<<<BT / 8, 256, 0, stream>>>(H1, ln3g, ln3b, ABF, BT, 256);
        gemm_big_bf16(ABF, wb[s][6], b3i, XG, BT, 768, 256);
        (void)hipMemsetAsync(HF, 0, (size_t)B_SZ * 256 * 4, stream);
        (void)hipMemsetAsync(HB, 0, (size_t)B_SZ * 256 * 2, stream);
        for (int t = 0; t < T_SZ; ++t) {
            gemm_sml_f32(HB, wb[s][7], b3h, GH, B_SZ, 768, 256);
            k_gru_gates<<<B_SZ * 256 / 256, 256, 0, stream>>>(
                XG, GH, HF, HB, H1, lens, t, T_SZ, 256, 256, 0, 1);
        }

        // ---- pooling into MLP input ----
        k_pool<<<B_SZ * 256 / 256, 256, 0, stream>>>(
            H1, lens, MLPI, T_SZ, 256, s * 512, 1024);
    }

    // ---- MLP head ----
    k_bn_lrelu_bf16<<<B_SZ * 1024 / 256, 256, 0, stream>>>(
        MLPI, (const float*)d_in[47], (const float*)d_in[48],
        (const float*)d_in[49], (const float*)d_in[50], MLPB, 1024);
    gemm_sml_f32(MLPB, mw1, (const float*)d_in[52], MLPF, B_SZ, 1024, 1024);
    k_bn_lrelu_bf16<<<B_SZ * 1024 / 256, 256, 0, stream>>>(
        MLPF, (const float*)d_in[53], (const float*)d_in[54],
        (const float*)d_in[55], (const float*)d_in[56], MLPB, 1024);
    gemm_sml_f32(MLPB, mw2, (const float*)d_in[58], MLPF, B_SZ, 512, 1024);
    k_bn_lrelu_bf16<<<B_SZ * 512 / 256, 256, 0, stream>>>(
        MLPF, (const float*)d_in[59], (const float*)d_in[60],
        (const float*)d_in[61], (const float*)d_in[62], MLPB, 512);
    gemm_sml_f32(MLPB, mw3, (const float*)d_in[64], MLPF, B_SZ, 256, 512);
    k_bn_lrelu_bf16<<<B_SZ * 256 / 256, 256, 0, stream>>>(
        MLPF, (const float*)d_in[65], (const float*)d_in[66],
        (const float*)d_in[67], (const float*)d_in[68], MLPB, 256);
    k_final<<<B_SZ, 16, 0, stream>>>(
        MLPB, (const float*)d_in[69], (const float*)d_in[70], (float*)d_out);
}
